// AttentionNet_14139032338898
// MI455X (gfx1250) — matrix operand order repaired and checked by reading, not compiled, not run on hardware
//
#include <hip/hip_runtime.h>
#include <hip/hip_bf16.h>
#include <math.h>

typedef __attribute__((ext_vector_type(16))) _Float16 v16h;
typedef __attribute__((ext_vector_type(8)))  float    v8f;

#define E_DIM   32
#define BB      2
#define RR      256
#define CC      1024
#define TPB     256
#define NWAVES  8
#define XSTR    40   // LDS row stride in halves (80B: 16B-multiple, conflict-free)

// ---------------------------------------------------------------------------
// Tensor Data Mover support (guarded: falls back to plain vector copy).
// ---------------------------------------------------------------------------
#if defined(__has_builtin)
#if __has_builtin(__builtin_amdgcn_tensor_load_to_lds) && \
    __has_builtin(__builtin_amdgcn_s_wait_tensorcnt)
#define HAVE_TDM 1
#endif
#endif

__device__ __forceinline__ unsigned lds_off(const void* p) {
  // AS(3) pointers are 32-bit offsets from the LDS base.
  return (unsigned)(unsigned long long)(__attribute__((address_space(3))) const char*)p;
}

#if defined(HAVE_TDM)
typedef __attribute__((ext_vector_type(4))) unsigned int v4u;
typedef __attribute__((ext_vector_type(8))) int v8i;
typedef __attribute__((ext_vector_type(4))) int v4i;

// 2D tile load: `rows` rows of 64 bytes (16 DW), global row pitch rowStrideDW
// (DWORD units), into LDS at ldsDst with +16B padding per row (pad_enable,
// pad_interval=3 -> every 16 DW, pad_amount=3 -> 4 DW) => 80B LDS row stride.
__device__ __forceinline__ void tdm_load_2d(const void* gsrc, unsigned ldsDst,
                                            unsigned rows,
                                            unsigned long long rowStrideDW) {
  const unsigned long long ga = (unsigned long long)gsrc;
  v4u g0;
  g0[0] = 1u;                                        // count=1 (valid user D#)
  g0[1] = ldsDst;                                    // lds_addr (bytes)
  g0[2] = (unsigned)(ga & 0xffffffffull);            // global_addr[31:0]
  g0[3] = (unsigned)((ga >> 32) & 0x01ffffffull)     // global_addr[56:32]
        | 0x80000000u;                               // type=2 ("image")
  v8i g1;
  g1[0] = (2 << 16)      // data_size = 4B
        | (1 << 20)      // pad_enable
        | (3 << 22)      // pad_interval: 16 DW
        | (3 << 25);     // pad_amount: 4 DW
  g1[1] = (int)(16u << 16);                          // tensor_dim0 = 16 DW
  g1[2] = (int)(rows << 16);                         // tensor_dim1 lo16
  g1[3] = (int)((rows >> 16) | (16u << 16));         // dim1 hi16 | tile_dim0=16
  g1[4] = (int)(rows & 0xffffu);                     // tile_dim1 (tile_dim2=0)
  g1[5] = (int)(rowStrideDW & 0xffffffffull);        // tensor_dim0_stride lo32
  g1[6] = (int)((rowStrideDW >> 32) & 0xffffull);    // stride hi16 (dim1_stride=0)
  g1[7] = 0;
  v4i gz = {0, 0, 0, 0};
#if __clang_major__ >= 23
  v8i gz8 = {0, 0, 0, 0, 0, 0, 0, 0};
  __builtin_amdgcn_tensor_load_to_lds(g0, g1, gz, gz, gz8, 0);
#else
  __builtin_amdgcn_tensor_load_to_lds(g0, g1, gz, gz, 0);
#endif
}
#endif  // HAVE_TDM

// ---------------------------------------------------------------------------
// WMMA fragment gathers: contiguous k = 16*(lane/16) + s in both A and B.
// ---------------------------------------------------------------------------
__device__ __forceinline__ v8f zero8() {
  v8f z;
#pragma unroll
  for (int i = 0; i < 8; ++i) z[i] = 0.0f;
  return z;
}

__device__ __forceinline__ v16h load_afrag(const _Float16* __restrict__ X, int stride,
                                           int r0, int k0, int lane) {
  const int m  = lane & 15;
  const int kb = (lane >> 4) << 4;
  const _Float16* row = X + (r0 + m) * stride + k0 + kb;
  v16h a;
#pragma unroll
  for (int i = 0; i < 8; ++i) a[i] = row[i];
#pragma unroll
  for (int i = 0; i < 8; ++i) a[8 + i] = row[8 + i];
  return a;
}

__device__ __forceinline__ v16h load_bfrag(const _Float16* __restrict__ W, int stride,
                                           int k0, int n0, int lane) {
  const int n  = lane & 15;
  const int ko = (lane >> 4) << 4;
  const _Float16* col = W + (k0 + ko) * stride + n0 + n;
  v16h b;
#pragma unroll
  for (int i = 0; i < 16; ++i) b[i] = col[i * stride];
  return b;
}

#define WMMA16(a, b, c) \
  __builtin_amdgcn_wmma_f32_16x16x32_f16(false, (a), false, (b), (short)0, (c), false, false)

__device__ __forceinline__ float elup1(float x) {      // elu(x)+1
  return x > 0.0f ? x + 1.0f : __expf(x);
}
__device__ __forceinline__ float gelu_exact(float x) { // approximate=False
  return 0.5f * x * (1.0f + erff(x * 0.70710678118654752f));
}

// ---------------------------------------------------------------------------
// Embed: one-hot conv + ReLU + channel LayerNorm; writes f16 activations.
// ---------------------------------------------------------------------------
__global__ __launch_bounds__(TPB) void embed_kernel(
    const int* __restrict__ x, const float* __restrict__ cw,
    const float* __restrict__ cb, const float* __restrict__ ng,
    const float* __restrict__ nb, _Float16* __restrict__ A, int total) {
  const int p = blockIdx.x * TPB + threadIdx.x;
  if (p >= total) return;
  const int xv = x[p];
  float v[32];
  float mean = 0.0f;
#pragma unroll
  for (int h = 0; h < 32; ++h) {
    float t = fmaxf(cw[h * 5 + xv] + cb[h], 0.0f);
    v[h] = t;
    mean += t;
  }
  mean *= (1.0f / 32.0f);
  float var = 0.0f;
#pragma unroll
  for (int h = 0; h < 32; ++h) { const float d = v[h] - mean; var += d * d; }
  var *= (1.0f / 32.0f);
  const float inv = rsqrtf(var + 1e-5f);
  _Float16* dst = A + (long long)p * 32;
#pragma unroll
  for (int h = 0; h < 32; ++h)
    dst[h] = (_Float16)((v[h] - mean) * inv * ng[h] + nb[h]);
}

// ---------------------------------------------------------------------------
// Linear attention over one sequence (row/col via strides, in half units).
// One workgroup per (b, outer); sequence staged to LDS by the TDM.
// ---------------------------------------------------------------------------
__global__ __launch_bounds__(TPB) void attn_kernel(
    _Float16* __restrict__ A,
    const float* __restrict__ qw, const float* __restrict__ qb,
    const float* __restrict__ kw, const float* __restrict__ kb,
    const float* __restrict__ vw, const float* __restrict__ vb,
    const float* __restrict__ ow, const float* __restrict__ ob,
    const float* __restrict__ lng, const float* __restrict__ lnb,
    int Lseq, long long strideSeq, long long strideOuter, int outerPerB,
    long long strideB) {
  extern __shared__ char smemRaw[];
  _Float16* Wq = (_Float16*)smemRaw;      // 4 x 1024 f16 weights
  _Float16* Wk = Wq + 1024;
  _Float16* Wv = Wk + 1024;
  _Float16* Wo = Wv + 1024;
  float* bias    = (float*)(Wo + 1024);   // qb kb vb ob lng lnb : 192 f32
  float* partial = bias + 192;            // 8 waves * 288 partials
  float* kvbuf   = partial + 8 * 288;     // KtV[h][d1][d2] (256) + ksum (32)
  float* wscr    = kvbuf + 288;           // per-wave scratch: 8 * (528+528) f32
  _Float16* Xh   = (_Float16*)(wscr + 8 * 1056); // Lseq rows, stride XSTR halves

  const int tid  = threadIdx.x;
  const int wave = tid >> 5;
  const int lane = tid & 31;

  const int g = blockIdx.x;
  const int b = g / outerPerB;
  const int o = g - b * outerPerB;
  const long long base = (long long)b * strideB + (long long)o * strideOuter;

  for (int i = tid; i < 1024; i += TPB) {
    Wq[i] = (_Float16)qw[i];
    Wk[i] = (_Float16)kw[i];
    Wv[i] = (_Float16)vw[i];
    Wo[i] = (_Float16)ow[i];
  }
  if (tid < 32)       bias[tid] = qb[tid];
  else if (tid < 64)  bias[tid] = kb[tid - 32];
  else if (tid < 96)  bias[tid] = vb[tid - 64];
  else if (tid < 128) bias[tid] = ob[tid - 96];
  else if (tid < 160) bias[tid] = lng[tid - 128];
  else if (tid < 192) bias[tid] = lnb[tid - 160];

  // Stage sequence into LDS: one TDM 2D descriptor (pad -> 80B LDS rows).
#if defined(HAVE_TDM)
  if (wave == 0) {
    tdm_load_2d((const void*)(A + base), lds_off(Xh), (unsigned)Lseq,
                (unsigned long long)(strideSeq >> 1));  // halves -> DWORDs
    __builtin_amdgcn_s_wait_tensorcnt(0);
  }
#else
  {
    const int nchunk = Lseq * 4;  // 16B chunks
    for (int v = tid; v < nchunk; v += TPB) {
      const int c = v >> 2, part = (v & 3) * 8;
      *(uint4*)(Xh + c * XSTR + part) =
          *(const uint4*)(A + base + (long long)c * strideSeq + part);
    }
  }
#endif
  __syncthreads();

  float* SA = wscr + wave * 1056;   // 16x33 f32
  float* SB = SA + 528;             // 16x33 f32 (aliased as f16 16x34)
  _Float16* SBH = (_Float16*)SB;

  const int nt = Lseq >> 4;         // 64 (row) or 16 (col): divisible by 8
  const int n  = lane & 15;
  const int mb = (lane >> 4) << 3;
  const int h  = lane >> 3;         // head owned by this lane (channel = lane)
  const int dd = lane & 7;

  // -------- Phase A: K/V projections, accumulate KtV and Ksum --------
  v16h bk0 = load_bfrag(Wk, 32, 0, 0, lane);
  v16h bk1 = load_bfrag(Wk, 32, 0, 16, lane);
  v16h bv0 = load_bfrag(Wv, 32, 0, 0, lane);
  v16h bv1 = load_bfrag(Wv, 32, 0, 16, lane);

  float kvacc[8];
#pragma unroll
  for (int i = 0; i < 8; ++i) kvacc[i] = 0.0f;
  float ksacc = 0.0f;

  for (int t = wave; t < nt; t += NWAVES) {
    const int r0 = t << 4;
    v16h a = load_afrag(Xh, XSTR, r0, 0, lane);
    v8f ck0 = WMMA16(a, bk0, zero8());
    v8f ck1 = WMMA16(a, bk1, zero8());
    v8f cv0 = WMMA16(a, bv0, zero8());
    v8f cv1 = WMMA16(a, bv1, zero8());
#pragma unroll
    for (int i = 0; i < 8; ++i) {
      SA[(mb + i) * 33 + n]      = elup1(ck0[i] + bias[32 + n]);
      SA[(mb + i) * 33 + 16 + n] = elup1(ck1[i] + bias[48 + n]);
      SB[(mb + i) * 33 + n]      = cv0[i] + bias[64 + n];
      SB[(mb + i) * 33 + 16 + n] = cv1[i] + bias[80 + n];
    }
    __builtin_amdgcn_wave_barrier();
    for (int cl = 0; cl < 16; ++cl) {
      const float kk1 = SA[cl * 33 + h * 8 + dd];   // K[c][h*8+d1]
      const float* vr = &SB[cl * 33 + h * 8];
#pragma unroll
      for (int d2 = 0; d2 < 8; ++d2) kvacc[d2] += kk1 * vr[d2];
      ksacc += SA[cl * 33 + lane];                  // ksum[e=lane]
    }
    __builtin_amdgcn_wave_barrier();
  }
  {
    float* myp = partial + wave * 288;
#pragma unroll
    for (int d2 = 0; d2 < 8; ++d2) myp[lane * 8 + d2] = kvacc[d2];
    myp[256 + lane] = ksacc;
  }
  __syncthreads();
  for (int idx = tid; idx < 288; idx += TPB) {   // deterministic fixed order
    float s = 0.0f;
#pragma unroll
    for (int w = 0; w < NWAVES; ++w) s += partial[w * 288 + idx];
    kvbuf[idx] = s;
  }
  __syncthreads();

  // -------- Phase B: Q projection, attention, O projection, LN --------
  v16h bq0 = load_bfrag(Wq, 32, 0, 0, lane);
  v16h bq1 = load_bfrag(Wq, 32, 0, 16, lane);
  v16h bo0 = load_bfrag(Wo, 32, 0, 0, lane);
  v16h bo1 = load_bfrag(Wo, 32, 0, 16, lane);

  for (int t = wave; t < nt; t += NWAVES) {
    const int r0 = t << 4;
    v16h a = load_afrag(Xh, XSTR, r0, 0, lane);
    v8f cq0 = WMMA16(a, bq0, zero8());
    v8f cq1 = WMMA16(a, bq1, zero8());
#pragma unroll
    for (int i = 0; i < 8; ++i) {
      SA[(mb + i) * 33 + n]      = elup1(cq0[i] + bias[n]);
      SA[(mb + i) * 33 + 16 + n] = elup1(cq1[i] + bias[16 + n]);
    }
    __builtin_amdgcn_wave_barrier();
    // lane owns channel e=lane: num = q . KtV[:,de], den = q . ksum
    for (int m = 0; m < 16; ++m) {
      const float* qrow = &SA[m * 33 + h * 8];
      const float* kvh  = &kvbuf[h * 64];
      const float* ksh  = &kvbuf[256 + h * 8];
      float num = 0.0f, den = 0.0f;
#pragma unroll
      for (int d = 0; d < 8; ++d) {
        const float qd = qrow[d];
        num += qd * kvh[d * 8 + dd];
        den += qd * ksh[d];
      }
      SBH[m * 34 + lane] = (_Float16)(num / (den + 1e-6f));
    }
    __builtin_amdgcn_wave_barrier();
    v16h av = load_afrag(SBH, 34, 0, 0, lane);
    v8f co0 = WMMA16(av, bo0, zero8());
    v8f co1 = WMMA16(av, bo1, zero8());
#pragma unroll
    for (int i = 0; i < 8; ++i) {
      const int m = mb + i;
      SA[m * 33 + n]      = co0[i] + bias[96 + n]  + (float)Xh[(r0 + m) * XSTR + n];
      SA[m * 33 + 16 + n] = co1[i] + bias[112 + n] + (float)Xh[(r0 + m) * XSTR + 16 + n];
    }
    __builtin_amdgcn_wave_barrier();
    {
      const int m  = lane >> 1;
      const int e0 = (lane & 1) << 4;
      float mean = 0.0f;
#pragma unroll
      for (int e = 0; e < 32; ++e) mean += SA[m * 33 + e];
      mean *= (1.0f / 32.0f);
      float var = 0.0f;
#pragma unroll
      for (int e = 0; e < 32; ++e) { const float d = SA[m * 33 + e] - mean; var += d * d; }
      var *= (1.0f / 32.0f);
      const float inv = rsqrtf(var + 1e-5f);
      _Float16* dst = A + base + (long long)(r0 + m) * strideSeq;
#pragma unroll
      for (int e = e0; e < e0 + 16; ++e)
        dst[e] = (_Float16)((SA[m * 33 + e] - mean) * inv * bias[128 + e] + bias[160 + e]);
    }
    __builtin_amdgcn_wave_barrier();
  }
}

// ---------------------------------------------------------------------------
// Pointwise FFN: 32 -> 128 (GELU) -> 32, residual, optional channel LN.
// One 16-position tile per wave; tile staged by a per-wave TDM descriptor.
// ---------------------------------------------------------------------------
__global__ __launch_bounds__(TPB) void ffn_kernel(
    _Float16* __restrict__ A,
    const float* __restrict__ w1, const float* __restrict__ b1,
    const float* __restrict__ w2, const float* __restrict__ b2,
    const float* __restrict__ lng, const float* __restrict__ lnb,
    int applyLN) {
  __shared__ __align__(16) _Float16 W1h[32 * 128];
  __shared__ __align__(16) _Float16 W2h[128 * 32];
  __shared__ float bias[224];  // b1[0:128] b2[128:160] lng[160:192] lnb[192:224]
  __shared__ __align__(16) _Float16 Xs[NWAVES][16 * XSTR];
  __shared__ __align__(16) _Float16 H1s[NWAVES][16 * 132];

  const int tid  = threadIdx.x;
  const int wave = tid >> 5;
  const int lane = tid & 31;

  for (int i = tid; i < 4096; i += TPB) {
    W1h[i] = (_Float16)w1[i];
    W2h[i] = (_Float16)w2[i];
  }
  if (tid < 128)      bias[tid] = b1[tid];
  else if (tid < 160) bias[tid] = b2[tid - 128];
  else if (tid < 192) bias[tid] = lng[tid - 160];
  else if (tid < 224) bias[tid] = lnb[tid - 192];
  __syncthreads();

  const long long T = (long long)blockIdx.x * NWAVES + wave;
  const long long gbase = T * (16 * 32);
  _Float16* xs = Xs[wave];
  _Float16* h1 = H1s[wave];

#if defined(HAVE_TDM)
  tdm_load_2d((const void*)(A + gbase), lds_off(xs), 16u, 16ull);
  __builtin_amdgcn_s_wait_tensorcnt(0);
#else
#pragma unroll
  for (int j = lane; j < 64; j += 32) {   // 64 chunks of 16B
    const int c = j >> 2, part = (j & 3) * 8;
    *(uint4*)(xs + c * XSTR + part) = *(const uint4*)(A + gbase + c * 32 + part);
  }
#endif
  __builtin_amdgcn_wave_barrier();

  const int n  = lane & 15;
  const int mb = (lane >> 4) << 3;

  v16h a = load_afrag(xs, XSTR, 0, 0, lane);
#pragma unroll
  for (int j = 0; j < 8; ++j) {
    v16h bw = load_bfrag(W1h, 128, 0, j * 16, lane);
    v8f c = WMMA16(a, bw, zero8());
#pragma unroll
    for (int i = 0; i < 8; ++i)
      h1[(mb + i) * 132 + j * 16 + n] = (_Float16)gelu_exact(c[i] + bias[j * 16 + n]);
  }
  __builtin_amdgcn_wave_barrier();

  v8f o0 = zero8(), o1 = zero8();
#pragma unroll
  for (int kk = 0; kk < 4; ++kk) {
    v16h a2  = load_afrag(h1, 132, 0, kk * 32, lane);
    v16h bw0 = load_bfrag(W2h, 32, kk * 32, 0, lane);
    v16h bw1 = load_bfrag(W2h, 32, kk * 32, 16, lane);
    o0 = WMMA16(a2, bw0, o0);   // chained K accumulation
    o1 = WMMA16(a2, bw1, o1);
  }
  __builtin_amdgcn_wave_barrier();
  float* SF = (float*)h1;  // reuse scratch: 16*33 f32 fits in 16*132 f16
#pragma unroll
  for (int i = 0; i < 8; ++i) {
    const int m = mb + i;
    SF[m * 33 + n]      = o0[i] + bias[128 + n]      + (float)xs[m * XSTR + n];
    SF[m * 33 + 16 + n] = o1[i] + bias[128 + 16 + n] + (float)xs[m * XSTR + 16 + n];
  }
  __builtin_amdgcn_wave_barrier();
  {
    const int m  = lane >> 1;
    const int e0 = (lane & 1) << 4;
    _Float16* dst = A + gbase + m * 32;
    if (applyLN) {
      float mean = 0.0f;
#pragma unroll
      for (int e = 0; e < 32; ++e) mean += SF[m * 33 + e];
      mean *= (1.0f / 32.0f);
      float var = 0.0f;
#pragma unroll
      for (int e = 0; e < 32; ++e) { const float d = SF[m * 33 + e] - mean; var += d * d; }
      var *= (1.0f / 32.0f);
      const float inv = rsqrtf(var + 1e-5f);
#pragma unroll
      for (int e = e0; e < e0 + 16; ++e)
        dst[e] = (_Float16)((SF[m * 33 + e] - mean) * inv * bias[160 + e] + bias[192 + e]);
    } else {
#pragma unroll
      for (int e = e0; e < e0 + 16; ++e) dst[e] = (_Float16)SF[m * 33 + e];
    }
  }
}

// ---------------------------------------------------------------------------
// Head: only (class=1, row=R-1) survives the squeezes; mean over C.
// ---------------------------------------------------------------------------
__global__ __launch_bounds__(TPB) void out_kernel(
    const _Float16* __restrict__ A, const float* __restrict__ ow,
    const float* __restrict__ ob, float* __restrict__ out) {
  __shared__ float red[TPB];
  __shared__ float wrow[32];
  const int b = blockIdx.x, tid = threadIdx.x;
  if (tid < 32) wrow[tid] = ow[32 + tid];  // out_w row for class 1
  __syncthreads();
  const _Float16* base = A + (((long long)b * RR + (RR - 1)) * CC) * 32;
  float acc = 0.0f;
  for (int c = tid; c < CC; c += TPB) {
    const _Float16* v = base + c * 32;
    float d = 0.0f;
#pragma unroll
    for (int e = 0; e < 32; ++e) d += (float)v[e] * wrow[e];
    acc += d;
  }
  red[tid] = acc;
  __syncthreads();
  for (int s = TPB / 2; s > 0; s >>= 1) {
    if (tid < s) red[tid] += red[tid + s];
    __syncthreads();
  }
  if (tid == 0) out[b] = red[0] * (1.0f / CC) + ob[1];
}

// ---------------------------------------------------------------------------
// Host launcher.  Param leaves in JAX sorted-key order; x located by its
// unique size so both "x first" and "x last" top-level flattening work.
// Per-block leaf layout (22): col{k_b,k_w,o_b,o_w,q_b,q_w,v_b,v_w}=0..7,
// ffn_b1=8, ffn_b2=9, ffn_w1=10, ffn_w2=11, ln_b=12, ln_g=13,
// row{k_b,k_w,o_b,o_w,q_b,q_w,v_b,v_w}=14..21.  After 88 block leaves:
// conv_in_b=88, conv_in_w=89, norm_b=90, norm_g=91, out_b=92, out_w=93.
// ---------------------------------------------------------------------------
extern "C" void kernel_launch(void* const* d_in, const int* in_sizes, int n_in,
                              void* d_out, int out_size, void* d_ws, size_t ws_size,
                              hipStream_t stream) {
  const int XTOT = BB * RR * CC;
  int xi = n_in - 1;
  for (int i = 0; i < n_in; ++i)
    if (in_sizes[i] == XTOT) { xi = i; break; }
  const int p0 = (xi == 0) ? 1 : 0;
  auto prm = [&](int i) -> const float* { return (const float*)d_in[p0 + i]; };
  const int* x = (const int*)d_in[xi];
  _Float16* A = (_Float16*)d_ws;  // 32 MB f16 activation tensor [B,R,C,32]
  if (ws_size < (size_t)XTOT * E_DIM * sizeof(_Float16)) return;

  embed_kernel<<<XTOT / TPB, TPB, 0, stream>>>(x, prm(89), prm(88), prm(91),
                                               prm(90), A, XTOT);

  constexpr size_t ATTN_FIXED =
      4 * 1024 * 2 + 192 * 4 + 8 * 288 * 4 + 288 * 4 + 8 * 1056 * 4;  // 53120 B
  const size_t shRow = ATTN_FIXED + (size_t)CC * XSTR * 2;  // ~132 KB
  const size_t shCol = ATTN_FIXED + (size_t)RR * XSTR * 2;  // ~72 KB

  for (int blk = 0; blk < 4; ++blk) {
    const int t = blk * 22;
    // Row attention: sequence along C.
    attn_kernel<<<BB * RR, TPB, shRow, stream>>>(
        A, prm(t + 19), prm(t + 18), prm(t + 15), prm(t + 14),
        prm(t + 21), prm(t + 20), prm(t + 17), prm(t + 16),
        prm(t + 13), prm(t + 12),
        CC, (long long)E_DIM, (long long)CC * E_DIM, RR,
        (long long)RR * CC * E_DIM);
    // Column attention: sequence along R.
    attn_kernel<<<BB * CC, TPB, shCol, stream>>>(
        A, prm(t + 5), prm(t + 4), prm(t + 1), prm(t + 0),
        prm(t + 7), prm(t + 6), prm(t + 3), prm(t + 2),
        prm(t + 13), prm(t + 12),
        RR, (long long)CC * E_DIM, (long long)E_DIM, CC,
        (long long)RR * CC * E_DIM);
    // FFN (+ LN except after last block).
    ffn_kernel<<<XTOT / 16 / NWAVES, TPB, 0, stream>>>(
        A, prm(t + 10), prm(t + 8), prm(t + 11), prm(t + 9),
        prm(t + 13), prm(t + 12), blk != 3 ? 1 : 0);
  }

  out_kernel<<<BB, TPB, 0, stream>>>(A, prm(93), prm(92), (float*)d_out);
}
